// BRTFusedExperts_44762149159150
// MI455X (gfx1250) — matrix operand order repaired and checked by reading, not compiled, not run on hardware
//
#include <hip/hip_runtime.h>

// MI455X / gfx1250, wave32. Per-expert MLP (no activation between fc1/fc2):
//   h   = x @ wi_w^T + wi_b     [E,64,8192]   (staged in d_ws as f16)
//   out = h @ wo_w^T + wo_b     [E,64,2048]
// HBM-bound on weight streaming (~1.07 GB @ 23.3 TB/s ~= 46 us floor).
// v_wmma_f32_16x16x32_f16 with explicit ping-pong register buffers: loads
// for k-step i+1 are in flight while WMMAs for k-step i execute, and no
// buffer-rotation copies are needed.

typedef __attribute__((ext_vector_type(16))) _Float16 v16h;
typedef __attribute__((ext_vector_type(8)))  _Float16 v8h;
typedef __attribute__((ext_vector_type(8)))  float    v8f;

// D[M=64, NTOT] = X[64, K] * W[NTOT, K]^T + bias; one expert per NBLK blocks.
// Workgroup: 256 threads = 8 waves; wave w -> m_tile = w&3, n_group = w>>2.
// Each wave computes a 16x64 strip: 4 accumulators of 16x16 f32.
template<int K, int NTOT>
__global__ __launch_bounds__(256) void wmma_linear_kernel(
    const _Float16* __restrict__ X,    // [E, 64, K]
    const _Float16* __restrict__ W,    // [E, NTOT, K]  (torch Linear layout)
    const _Float16* __restrict__ Bv,   // [E, NTOT]
    _Float16* __restrict__ Out)        // [E, 64, NTOT]
{
    constexpr int M     = 64;
    constexpr int NTILE = 128;           // N columns per workgroup
    constexpr int NBLK  = NTOT / NTILE;  // blocks per expert

    const int e    = blockIdx.x / NBLK;
    const int nb   = blockIdx.x - e * NBLK;

    const int lane = threadIdx.x & 31;
    const int wave = threadIdx.x >> 5;
    const int mt   = wave & 3;           // m tile: rows mt*16 .. +15
    const int ng   = wave >> 2;          // n group: cols ng*64 .. +63
    const int l15  = lane & 15;
    const int kq   = lane >> 4;          // 0 or 1 (K-segment select)

    // A fragment (16x32 f16): lane row = mt*16 + l15.
    //   VGPR0-3 <- K[k0 + kq*16 .. +7], VGPR4-7 <- K[k0 + kq*16 + 8 .. +7]
    const _Float16* Xrow =
        X + ((size_t)e * M + (size_t)(mt * 16 + l15)) * K + kq * 16;

    // B fragment (32x16 f16, B = W^T): lane col n = n0 + t*16 + l15,
    //   16 contiguous K-halves starting at k0 + kq*16 -> one 32B load.
    const int n0 = nb * NTILE + ng * 64;
    const _Float16* Wp[4];
#pragma unroll
    for (int t = 0; t < 4; ++t) {
        const int n = n0 + t * 16 + l15;
        Wp[t] = W + ((size_t)e * NTOT + n) * K + kq * 16;
    }

    auto loadA = [&](int k0) -> v16h {
        v8h lo = *(const v8h*)(Xrow + k0);
        v8h hi = *(const v8h*)(Xrow + k0 + 8);
        v16h a;
#pragma unroll
        for (int i = 0; i < 8; ++i) { a[i] = lo[i]; a[i + 8] = hi[i]; }
        return a;
    };

    v8f acc[4] = {};

    // Ping-pong fragment buffers; no rotation copies in source.
    v16h a0, a1, b0[4], b1[4];

    // prologue: fill buffer 0 with k-step 0
    a0 = loadA(0);
#pragma unroll
    for (int t = 0; t < 4; ++t) b0[t] = *(const v16h*)(Wp[t]);

    // steady state: each trip consumes two k-steps (buf0 then buf1),
    // prefetching the next step into the other buffer before computing.
    int k0 = 0;
    for (; k0 < K - 64; k0 += 64) {
        // fill buf1 with k-step (k0+32), then compute k-step k0 from buf0
        a1 = loadA(k0 + 32);
#pragma unroll
        for (int t = 0; t < 4; ++t) b1[t] = *(const v16h*)(Wp[t] + k0 + 32);
#pragma unroll
        for (int t = 0; t < 4; ++t) {
            acc[t] = __builtin_amdgcn_wmma_f32_16x16x32_f16(
                false, a0, false, b0[t], (short)0, acc[t], false, false);
        }

        // fill buf0 with k-step (k0+64), then compute k-step (k0+32) from buf1
        a0 = loadA(k0 + 64);
#pragma unroll
        for (int t = 0; t < 4; ++t) b0[t] = *(const v16h*)(Wp[t] + k0 + 64);
#pragma unroll
        for (int t = 0; t < 4; ++t) {
            acc[t] = __builtin_amdgcn_wmma_f32_16x16x32_f16(
                false, a1, false, b1[t], (short)0, acc[t], false, false);
        }
    }

    // pipeline drain: buf0 holds k-step K-64; fetch last step into buf1
    a1 = loadA(K - 32);
#pragma unroll
    for (int t = 0; t < 4; ++t) b1[t] = *(const v16h*)(Wp[t] + K - 32);
#pragma unroll
    for (int t = 0; t < 4; ++t) {
        acc[t] = __builtin_amdgcn_wmma_f32_16x16x32_f16(
            false, a0, false, b0[t], (short)0, acc[t], false, false);
    }
#pragma unroll
    for (int t = 0; t < 4; ++t) {
        acc[t] = __builtin_amdgcn_wmma_f32_16x16x32_f16(
            false, a1, false, b1[t], (short)0, acc[t], false, false);
    }

    // Bias add + f16 store.
    // D layout: VGPR r -> row (mt*16 + r + 8*kq), col (n&15 across lanes).
    const _Float16* bp = Bv + (size_t)e * NTOT + n0 + l15;
    _Float16* outBase  = Out + (size_t)e * M * NTOT + n0 + l15;
#pragma unroll
    for (int t = 0; t < 4; ++t) {
        const float bf = (float)bp[t * 16];
        _Float16* op = outBase + t * 16
                     + (size_t)(mt * 16 + kq * 8) * NTOT;
#pragma unroll
        for (int r = 0; r < 8; ++r) {
            op[(size_t)r * NTOT] = (_Float16)(acc[t][r] + bf);
        }
    }
}

extern "C" void kernel_launch(void* const* d_in, const int* in_sizes, int n_in,
                              void* d_out, int out_size, void* d_ws, size_t ws_size,
                              hipStream_t stream) {
    (void)in_sizes; (void)n_in; (void)out_size; (void)ws_size;

    constexpr int E = 16, H = 2048, I = 8192;

    const _Float16* x    = (const _Float16*)d_in[0]; // [E,1,64,H]
    const _Float16* wi_w = (const _Float16*)d_in[1]; // [E,I,H]
    const _Float16* wi_b = (const _Float16*)d_in[2]; // [E,I]
    const _Float16* wo_w = (const _Float16*)d_in[3]; // [E,H,I]
    const _Float16* wo_b = (const _Float16*)d_in[4]; // [E,H]

    _Float16* h   = (_Float16*)d_ws;    // [E,64,I] f16 = 16 MiB scratch
    _Float16* out = (_Float16*)d_out;   // [E,1,64,H] f16 (reference dtype)

    dim3 blk(256);
    // fc1: K=H, N=I -> 16 * 64 = 1024 workgroups
    wmma_linear_kernel<H, I><<<dim3(E * (I / 128)), blk, 0, stream>>>(
        x, wi_w, wi_b, h);
    // fc2: K=I, N=H -> 16 * 16 = 256 workgroups (stream-ordered after fc1)
    wmma_linear_kernel<I, H><<<dim3(E * (H / 128)), blk, 0, stream>>>(
        h, wo_w, wo_b, out);
}
